// CompressionEncoder_52570399703654
// MI455X (gfx1250) — matrix operand order repaired and checked by reading, compiled, not run on hardware
//
#include <hip/hip_runtime.h>
#include <hip/hip_bf16.h>

typedef __attribute__((ext_vector_type(16))) __bf16 bf16x16;
typedef __attribute__((ext_vector_type(8)))  float  v8f;
typedef __attribute__((ext_vector_type(4)))  unsigned tdm_v4u;
typedef __attribute__((ext_vector_type(8)))  int      tdm_v8i;
typedef __attribute__((ext_vector_type(4)))  int      tdm_v4i;

#if __has_builtin(__builtin_amdgcn_tensor_load_to_lds) && \
    __has_builtin(__builtin_amdgcn_s_wait_tensorcnt)
#define HAVE_TDM 1
#endif

// ---------------------------------------------------------------- utilities
__global__ __launch_bounds__(256) void zero_f32(float* p, int n) {
    int i = blockIdx.x * 256 + threadIdx.x;
    if (i < n) p[i] = 0.0f;
}

__global__ __launch_bounds__(256) void cvt_f32_bf16(const float* __restrict__ s,
                                                    __bf16* __restrict__ d, int n) {
    int i = blockIdx.x * 256 + threadIdx.x;
    if (i < n) d[i] = (__bf16)s[i];
}

// ---------------------------------------------------------------- conv1
// x[16,1,512,512] f32 -> h1[16,64,256,256] bf16, 4x4 s2 p1 + bias + relu
__global__ __launch_bounds__(256) void conv1_relu_kernel(
    const float* __restrict__ x, const float* __restrict__ w1,
    const float* __restrict__ b1, __bf16* __restrict__ h1) {
    __shared__ float wsh[64 * 16];
    __shared__ float bsh[64];
    const int tid = threadIdx.x;
    for (int i = tid; i < 64 * 16; i += 256) wsh[i] = w1[i];
    if (tid < 64) bsh[tid] = b1[tid];
    __syncthreads();

    const int idx = blockIdx.x * 256 + tid;       // 16*256*256 threads
    const int wo = idx & 255;
    const int ho = (idx >> 8) & 255;
    const int n  = idx >> 16;

    float win[16];
    const int h0 = 2 * ho - 1, w0 = 2 * wo - 1;
#pragma unroll
    for (int kh = 0; kh < 4; ++kh) {
#pragma unroll
        for (int kw = 0; kw < 4; ++kw) {
            int h = h0 + kh, w = w0 + kw;
            float v = 0.0f;
            if (h >= 0 && h < 512 && w >= 0 && w < 512)
                v = x[((size_t)n * 512 + h) * 512 + w];
            win[kh * 4 + kw] = v;
        }
    }
    for (int oc = 0; oc < 64; ++oc) {
        float acc = bsh[oc];
        const float* wr = &wsh[oc * 16];
#pragma unroll
        for (int k = 0; k < 16; ++k) acc = fmaf(wr[k], win[k], acc);
        acc = acc > 0.0f ? acc : 0.0f;
        h1[(((size_t)n * 64 + oc) * 256 + ho) * 256 + wo] = (__bf16)acc;
    }
}

// ---------------------------------------------------------------- conv2/3
// Implicit-GEMM bf16 WMMA: 4x4 stride-2 pad-1 conv, Cout=128, relu, bf16 out.
// Block = 256 threads (8 waves). Block tile: (n, ho, 16 consecutive wo) x 128 Cout.
// Wave w handles Cout rows [16w, 16w+16). Full-K im2col strip staged in LDS.
// Interior tiles stage via the Tensor Data Mover (tensor_load_to_lds): the strip
// is a dense 3D tile x=36 elems, y=4 rows (stride Win), z=Cin ch (stride Hin*Win),
// and with tile_dim0==36 the TDM's linear LDS fill matches the COLS=36 layout.
template <int Cin, int Cout, int Hin, int Win>
__global__ __launch_bounds__(256) void conv4x4s2_wmma_kernel(
    const __bf16* __restrict__ in, const __bf16* __restrict__ wB,
    const float* __restrict__ bias, __bf16* __restrict__ out) {
    constexpr int Hout = Hin / 2, Wout = Win / 2;
    constexpr int K = Cin * 16;
    constexpr int COLS = 36;                         // strip stride 36 (72B, 4B-aligned)
    __shared__ __bf16 lds[Cin * 4 * COLS];

    const int wt  = blockIdx.x;                      // w tile
    const int ho  = blockIdx.y;
    const int n   = blockIdx.z;
    const int tid = threadIdx.x;
    const int w0  = wt * 16;
    const int hin0 = 2 * ho - 1;
    const int win0 = 2 * w0 - 1;

    const bool interior = (hin0 >= 0) && (hin0 + 4 <= Hin) &&
                          (win0 >= 0) && (win0 + COLS <= Win);

#if defined(HAVE_TDM)
    if (interior) {
        // one TDM descriptor per workgroup, issued by wave 0 only
        if (tid < 32) {
            const __bf16* gsrc =
                in + (((size_t)n * Cin) * Hin + hin0) * Win + win0;
            unsigned long long ga = (unsigned long long)(size_t)gsrc;
            unsigned lds_byte = (unsigned)(size_t)(&lds[0]);   // flat[31:0]==LDS offset

            tdm_v4u g0;
            g0[0] = 1u;                                        // count=1 (valid), user D#
            g0[1] = lds_byte;                                  // lds_addr
            g0[2] = (unsigned)(ga & 0xFFFFFFFFu);              // global_addr[31:0]
            g0[3] = (unsigned)((ga >> 32) & 0x01FFFFFFu)       // global_addr[56:32]
                    | 0x80000000u;                             // type=2 (bits127:126=10b)

            const unsigned td0 = (unsigned)(Win - win0);       // remaining x extent
            const unsigned td1 = (unsigned)(Hin - hin0);       // remaining y extent
            const unsigned long long s0 = (unsigned long long)Win;        // y stride
            const unsigned long long s1 = (unsigned long long)Hin * Win;  // z stride

            tdm_v8i g1;
            g1[0] = 0x00010000;                                // data_size=1 (2 bytes)
            g1[1] = (int)((td0 & 0xFFFFu) << 16);              // tensor_dim0[15:0]
            g1[2] = (int)(((td0 >> 16) & 0xFFFFu) | ((td1 & 0xFFFFu) << 16));
            g1[3] = (int)(((td1 >> 16) & 0xFFFFu) | ((unsigned)COLS << 16)); // tile_dim0
            g1[4] = (int)(4u | ((unsigned)Cin << 16));         // tile_dim1=4, tile_dim2=Cin
            g1[5] = (int)(unsigned)(s0 & 0xFFFFFFFFu);         // dim0_stride[31:0]
            g1[6] = (int)(((s0 >> 32) & 0xFFFFu) | ((unsigned)(s1 & 0xFFFFu) << 16));
            g1[7] = (int)(unsigned)((s1 >> 16) & 0xFFFFFFFFu); // dim1_stride[47:16]

            tdm_v4i g2;
            g2[0] = (int)Cin;                                  // tensor_dim2
            g2[1] = 0; g2[2] = 0; g2[3] = 0;                   // no dim3 / no iterate
            tdm_v4i g3; g3[0] = 0; g3[1] = 0; g3[2] = 0; g3[3] = 0;

#if defined(__clang_major__) && (__clang_major__ >= 23)
            tdm_v8i gpad; gpad[0]=0; gpad[1]=0; gpad[2]=0; gpad[3]=0;
            gpad[4]=0; gpad[5]=0; gpad[6]=0; gpad[7]=0;
            __builtin_amdgcn_tensor_load_to_lds(g0, g1, g2, g3, gpad, 0);
#else
            __builtin_amdgcn_tensor_load_to_lds(g0, g1, g2, g3, 0);
#endif
            __builtin_amdgcn_s_wait_tensorcnt(0);
        }
    } else
#endif
    {
        // scalar staging (edge tiles, or no-TDM toolchain): zero-pad halo
        for (int i = tid; i < Cin * 4 * COLS; i += 256) {
            int ci  = i / (4 * COLS);
            int rem = i % (4 * COLS);
            int kh  = rem / COLS;
            int c   = rem % COLS;
            int h = hin0 + kh, w = win0 + c;
            __bf16 v = (__bf16)0.0f;
            if (c < 35 && h >= 0 && h < Hin && w >= 0 && w < Win)
                v = in[(((size_t)n * Cin + ci) * Hin + h) * Win + w];
            lds[i] = v;
        }
    }
    __syncthreads();

    const int lane = tid & 31;
    const int wave = tid >> 5;
    const int coutBase = wave * 16;
    const int mrow = lane & 15;
    const int ncol = lane & 15;
    const bool hi  = lane >= 16;

    v8f acc = {};
    const __bf16* wRow = wB + (size_t)(coutBase + mrow) * K;

    for (int it = 0; it < Cin / 2; ++it) {           // K chunks of 32 (2 input channels)
        const int kc = it * 32;
        // A-matrix: 16x32 bf16 weights. Lane<16: K 0..15; lane>=16: K 16..31.
        const int kA = kc + (hi ? 16 : 0);
        if (it + 2 < Cin / 2) __builtin_prefetch(wRow + kA + 64, 0, 1);
        union { uint4 q[2]; bf16x16 v; } ua;
        ua.q[0] = *reinterpret_cast<const uint4*>(wRow + kA);
        ua.q[1] = *reinterpret_cast<const uint4*>(wRow + kA + 8);
        // B-matrix: 32x16 im2col from LDS. Lane<16: K 0..15; lane>=16: K 16..31; N = lane&15.
        union { unsigned u[8]; bf16x16 v; } ub;
        const int kB = hi ? 16 : 0;
#pragma unroll
        for (int j = 0; j < 8; ++j) {
            int kk  = kB + 2 * j;                    // chunk-local even K
            int ciL = kk >> 4;
            int tap = kk & 15;
            int kh  = tap >> 2, kw = tap & 3;        // kw even -> 4B-aligned LDS read
            int off = ((2 * it + ciL) * 4 + kh) * COLS + (2 * ncol + kw);
            ub.u[j] = *reinterpret_cast<const unsigned*>(&lds[off]);
        }
        acc = __builtin_amdgcn_wmma_f32_16x16x32_bf16(
            /*neg_a=*/false, ua.v, /*neg_b=*/false, ub.v,
            /*c_mod=*/(short)0, acc, /*reuse_a=*/false, /*reuse_b=*/false);
    }

    // --- epilogue: bias + relu, C/D layout (lanes 0-15: M=r, lanes 16-31: M=r+8)
#pragma unroll
    for (int r = 0; r < 8; ++r) {
        int m = coutBase + r + (hi ? 8 : 0);
        float v = acc[r] + bias[m];
        v = v > 0.0f ? v : 0.0f;
        out[(((size_t)n * Cout + m) * Hout + ho) * Wout + (w0 + ncol)] = (__bf16)v;
    }
}

// ---------------------------------------------------------------- conv4 + VQ
// h3[16,128,64,64] bf16 -> z[16,4,64,64] (3x3 s1 p1, Cout=4), then per row of 4
// horizontally-adjacent pixels: nearest of 512 dim-4 codes. Writes z_q (==z_q_st),
// idx, and accumulates sum((z_q - z)^2).
__global__ __launch_bounds__(256) void conv4_vq_kernel(
    const __bf16* __restrict__ h3, const __bf16* __restrict__ w4b,
    const float* __restrict__ b4, const float* __restrict__ emb,
    float* __restrict__ outZ, int* __restrict__ outIdx,
    float* __restrict__ lossAcc) {
    __shared__ float embs[512 * 4];
    __shared__ __bf16 w4s[4 * 1152];
    __shared__ float red[256];
    const int tid = threadIdx.x;
    for (int i = tid; i < 512 * 4; i += 256) embs[i] = emb[i];
    for (int i = tid; i < 4 * 1152; i += 256) w4s[i] = w4b[i];
    __syncthreads();

    const int row = blockIdx.x * 256 + tid;          // 65536 rows
    const int j = row & 15;                          // 4-pixel group in W
    const int h = (row >> 4) & 63;
    const int c = (row >> 10) & 3;
    const int n = row >> 12;

    float z[4] = {b4[c], b4[c], b4[c], b4[c]};
    const int wb = 4 * j - 1;
    const __bf16* wq = &w4s[c * 1152];
    for (int ci = 0; ci < 128; ++ci) {
        float wv[18];
#pragma unroll
        for (int kh = 0; kh < 3; ++kh) {
            int hh = h - 1 + kh;
#pragma unroll
            for (int cc = 0; cc < 6; ++cc) {
                int ww = wb + cc;
                float v = 0.0f;
                if (hh >= 0 && hh < 64 && ww >= 0 && ww < 64)
                    v = (float)h3[(((size_t)n * 128 + ci) * 64 + hh) * 64 + ww];
                wv[kh * 6 + cc] = v;
            }
        }
        const __bf16* wc = wq + ci * 9;
#pragma unroll
        for (int kh = 0; kh < 3; ++kh)
#pragma unroll
            for (int kw = 0; kw < 3; ++kw) {
                float wgt = (float)wc[kh * 3 + kw];
#pragma unroll
                for (int t = 0; t < 4; ++t)
                    z[t] = fmaf(wgt, wv[kh * 6 + kw + t], z[t]);
            }
    }

    // nearest codebook entry
    int best = 0;
    float bd = 3.4e38f;
    for (int e = 0; e < 512; ++e) {
        const float* ev = &embs[e * 4];
        float d = 0.0f;
#pragma unroll
        for (int t = 0; t < 4; ++t) {
            float df = z[t] - ev[t];
            d = fmaf(df, df, d);
        }
        if (d < bd) { bd = d; best = e; }
    }

    // z_q_st == z_q numerically; rows are contiguous in NCHW flat order.
    const float* ev = &embs[best * 4];
    float4 q = make_float4(ev[0], ev[1], ev[2], ev[3]);
    *reinterpret_cast<float4*>(outZ + (size_t)4 * row) = q;
    outIdx[row] = best;

    // loss partial: bd == sum_t (z_q - z)^2
    red[tid] = bd;
    __syncthreads();
    for (int s = 128; s > 0; s >>= 1) {
        if (tid < s) red[tid] += red[tid + s];
        __syncthreads();
    }
    if (tid == 0) atomicAdd(lossAcc, red[0]);
}

__global__ void finalize_loss_kernel(const float* lossAcc, float* outLoss) {
    // loss = mean((zq-z)^2) + 0.25*mean((zq-z)^2) = 1.25 * sum / 262144
    *outLoss = *lossAcc * (1.25f / 262144.0f);
}

// ---------------------------------------------------------------- launch
extern "C" void kernel_launch(void* const* d_in, const int* in_sizes, int n_in,
                              void* d_out, int out_size, void* d_ws, size_t ws_size,
                              hipStream_t stream) {
    (void)in_sizes; (void)n_in; (void)out_size; (void)ws_size;
    const float* x   = (const float*)d_in[0];
    const float* w1  = (const float*)d_in[1];
    const float* b1  = (const float*)d_in[2];
    const float* w2  = (const float*)d_in[3];
    const float* b2  = (const float*)d_in[4];
    const float* w3  = (const float*)d_in[5];
    const float* b3  = (const float*)d_in[6];
    const float* w4  = (const float*)d_in[7];
    const float* b4  = (const float*)d_in[8];
    const float* emb = (const float*)d_in[9];

    // workspace layout (bytes)
    char* ws = (char*)d_ws;
    constexpr size_t H1_E = (size_t)16 * 64 * 256 * 256;    // bf16
    constexpr size_t H2_E = (size_t)16 * 128 * 128 * 128;   // bf16
    constexpr size_t H3_E = (size_t)16 * 128 * 64 * 64;     // bf16
    constexpr size_t W2_E = 128 * 64 * 16;
    constexpr size_t W3_E = 128 * 128 * 16;
    constexpr size_t W4_E = 4 * 128 * 9;
    size_t off = 0;
    __bf16* h1  = (__bf16*)(ws + off); off += H1_E * 2;
    __bf16* h2  = (__bf16*)(ws + off); off += H2_E * 2;
    __bf16* h3  = (__bf16*)(ws + off); off += H3_E * 2;
    __bf16* w2b = (__bf16*)(ws + off); off += W2_E * 2;
    __bf16* w3b = (__bf16*)(ws + off); off += W3_E * 2;
    __bf16* w4b = (__bf16*)(ws + off); off += ((W4_E * 2 + 255) / 256) * 256;
    float*  lossAcc = (float*)(ws + off);

    float* outZ    = (float*)d_out;                  // [16,4,64,64] = 262144
    float* outLoss = outZ + 262144;                  // [1]
    int*   outIdx  = (int*)(outZ + 262145);          // [65536] int32 bit patterns

    zero_f32<<<1, 256, 0, stream>>>(lossAcc, 1);
    cvt_f32_bf16<<<(W2_E + 255) / 256, 256, 0, stream>>>(w2, w2b, (int)W2_E);
    cvt_f32_bf16<<<(W3_E + 255) / 256, 256, 0, stream>>>(w3, w3b, (int)W3_E);
    cvt_f32_bf16<<<(W4_E + 255) / 256, 256, 0, stream>>>(w4, w4b, (int)W4_E);

    conv1_relu_kernel<<<16 * 256 * 256 / 256, 256, 0, stream>>>(x, w1, b1, h1);

    conv4x4s2_wmma_kernel<64, 128, 256, 256>
        <<<dim3(128 / 16, 128, 16), 256, 0, stream>>>(h1, w2b, b2, h2);
    conv4x4s2_wmma_kernel<128, 128, 128, 128>
        <<<dim3(64 / 16, 64, 16), 256, 0, stream>>>(h2, w3b, b3, h3);

    conv4_vq_kernel<<<65536 / 256, 256, 0, stream>>>(h3, w4b, b4, emb,
                                                     outZ, outIdx, lossAcc);
    finalize_loss_kernel<<<1, 1, 0, stream>>>(lossAcc, outLoss);
}
